// Network_38491496907327
// MI455X (gfx1250) — matrix operand order repaired and checked by reading, not compiled, not run on hardware
//
#include <hip/hip_runtime.h>

typedef __attribute__((ext_vector_type(16))) __bf16 v16bf;
typedef __attribute__((ext_vector_type(8)))  __bf16 v8bf;
typedef __attribute__((ext_vector_type(4)))  __bf16 v4bf;
typedef __attribute__((ext_vector_type(8)))  float  v8f;
typedef __attribute__((ext_vector_type(4)))  float  v4f;

#define KDIM   512
#define NROWS  65536
#define RPB    64                 // rows per block
#define NBLK   (NROWS / RPB)      // 1024
#define XBS    520                // bf16 LDS row stride (4-bank step/row, conflict-free A loads)
#define XFS    516                // f32  LDS row stride (4-bank step/row, conflict-free dot loads)
#define BN_EPS 1e-5f

// ---------------- k0: C1 (f32, row-major) -> C1T (bf16, transposed) ----------
__global__ void k0_convert_transpose(const float* __restrict__ C1,
                                     __bf16* __restrict__ C1T) {
  const int j = blockIdx.x;                       // output row = C1 column
  for (int k = threadIdx.x; k < KDIM; k += blockDim.x)
    C1T[(size_t)j * KDIM + k] = (__bf16)C1[(size_t)k * KDIM + j];
}

// ---------------- k1: quad via WMMA, fused y + per-block BN partials ---------
__global__ __launch_bounds__(256)
void k1_quad_y_stats(const float* __restrict__ X,
                     const __bf16* __restrict__ C1T,
                     const float* __restrict__ C2,
                     const float* __restrict__ C3,
                     float* __restrict__ Y,
                     float* __restrict__ psum,
                     float* __restrict__ psum2) {
  __shared__ float  sXf[RPB * XFS];   // 132,096 B f32 X tile (async-copied from HBM)
  __shared__ __bf16 sXb[RPB * XBS];   //  66,560 B bf16 X tile (A fragments)
  __shared__ float  sQuad[RPB];

  const int t    = threadIdx.x;
  const int b    = blockIdx.x;
  const int row0 = b * RPB;

  if (t < RPB) sQuad[t] = 0.0f;

  // Phase A1: async copy the 64x512 f32 tile HBM -> LDS (padded rows),
  // 16B per lane per issue, 32 issues per thread. Tracked by ASYNCcnt.
  {
    const float* gbase = X + (size_t)row0 * KDIM;
    for (int i = 0; i < 32; ++i) {
      const int c  = t + 256 * i;       // 16B chunk index, 8192 total
      const int r  = c >> 7;            // row (128 chunks per row)
      const int ch = c & 127;
      const float*  gaddr   = gbase + (size_t)r * KDIM + ch * 4;
      unsigned      lds_off = (unsigned)(size_t)(void*)&sXf[r * XFS + ch * 4];
      asm volatile("global_load_async_to_lds_b128 %0, %1, off"
                   :: "v"(lds_off), "v"(gaddr) : "memory");
    }
    asm volatile("s_wait_asynccnt 0x0" ::: "memory");
  }
  __syncthreads();

  // Phase A2: convert f32 tile (LDS) -> bf16 tile (LDS)
  for (int e = t; e < RPB * (KDIM / 4); e += 256) {
    const int r  = e >> 7;
    const int c4 = e & 127;
    v4f x = *(const v4f*)(&sXf[r * XFS + c4 * 4]);
    v4bf bv;
    bv[0] = (__bf16)x[0]; bv[1] = (__bf16)x[1];
    bv[2] = (__bf16)x[2]; bv[3] = (__bf16)x[3];
    *(v4bf*)(&sXb[r * XBS + c4 * 4]) = bv;
  }
  __syncthreads();

  // Phase B: each wave owns 4 column-tiles, processed as 2 passes of 2.
  // Per K-step: 4 A fragments (LDS) + 2 B fragments (global) -> 8 WMMAs.
  const int wave = t >> 5, lane = t & 31;
  const int jloc = lane & 15;
  const int hs   = lane >> 4;         // lane half select

  float qacc[32];                     // [rt][i] quad partials
  #pragma unroll
  for (int i = 0; i < 32; ++i) qacc[i] = 0.f;

  #pragma unroll 1
  for (int pass = 0; pass < 2; ++pass) {
    // Opaque memory barrier: A-fragment LDS loads are invariant across passes;
    // without this LICM hoists all 64 fragments and spills them to scratch.
    asm volatile("" ::: "memory");

    const int ct0 = wave * 4 + pass * 2;
    const int ct1 = ct0 + 1;
    // B fragments: C1T row j, 16 consecutive K (lanes>=16 offset +16)
    const __bf16* b0p = C1T + (size_t)(ct0 * 16 + jloc) * KDIM + hs * 16;
    const __bf16* b1p = C1T + (size_t)(ct1 * 16 + jloc) * KDIM + hs * 16;

    v8f acc[8];                       // acc[rt*2 + c]
    #pragma unroll
    for (int i = 0; i < 8; ++i) acc[i] = (v8f){};

    #pragma unroll
    for (int ks = 0; ks < 16; ++ks) {
      const int kb = ks * 32;
      v16bf B0 = *(const v16bf*)(b0p + kb);
      v16bf B1 = *(const v16bf*)(b1p + kb);
      v16bf A[4];
      #pragma unroll
      for (int rt = 0; rt < 4; ++rt) {
        // A layout: lane row = rt*16 + lane%16; 16 consecutive K (lanes>=16 offset +16)
        const __bf16* aBase = &sXb[(rt * 16 + jloc) * XBS + hs * 16 + kb];
        v8bf alo = *(const v8bf*)(aBase);
        v8bf ahi = *(const v8bf*)(aBase + 8);
        A[rt] = __builtin_shufflevector(alo, ahi,
                  0,1,2,3,4,5,6,7,8,9,10,11,12,13,14,15);
      }
      #pragma unroll
      for (int rt = 0; rt < 4; ++rt) {
        acc[rt * 2 + 0] = __builtin_amdgcn_wmma_f32_16x16x32_bf16(
            false, A[rt], false, B0, (short)0, acc[rt * 2 + 0], false, false);
        acc[rt * 2 + 1] = __builtin_amdgcn_wmma_f32_16x16x32_bf16(
            false, A[rt], false, B1, (short)0, acc[rt * 2 + 1], false, false);
      }
    }

    // dot accumulator tiles vs f32 X in LDS (C/D layout: row = i + 8*hs, col = lane%16)
    #pragma unroll
    for (int rt = 0; rt < 4; ++rt) {
      const int xb0 = (rt * 16 + hs * 8) * XFS + ct0 * 16 + jloc;
      const int xb1 = (rt * 16 + hs * 8) * XFS + ct1 * 16 + jloc;
      #pragma unroll
      for (int i = 0; i < 8; ++i) {
        qacc[rt * 8 + i] += acc[rt * 2 + 0][i] * sXf[xb0 + i * XFS];
        qacc[rt * 8 + i] += acc[rt * 2 + 1][i] * sXf[xb1 + i * XFS];
      }
    }
  }

  #pragma unroll
  for (int rt = 0; rt < 4; ++rt)
    #pragma unroll
    for (int i = 0; i < 8; ++i)
      atomicAdd(&sQuad[rt * 16 + hs * 8 + i], qacc[rt * 8 + i]);   // ds_add_f32
  __syncthreads();

  // Phase C: y = quad + C2*x + C3; thread owns columns t and t+256, x from LDS
  const float c3v = C3[0];
  const int   c0 = t, c1 = t + 256;
  const float w0 = C2[c0], w1 = C2[c1];
  float s0 = 0.f, q0 = 0.f, s1 = 0.f, q1 = 0.f;
  for (int r = 0; r < RPB; ++r) {
    const float  qd   = sQuad[r];
    const float  xa   = sXf[r * XFS + c0];
    const float  xb   = sXf[r * XFS + c1];
    const float  ya   = qd + w0 * xa + c3v;
    const float  yb   = qd + w1 * xb + c3v;
    const size_t base = (size_t)(row0 + r) * KDIM;
    Y[base + c0] = ya; Y[base + c1] = yb;
    s0 += ya; q0 += ya * ya;
    s1 += yb; q1 += yb * yb;
  }
  psum [(size_t)b * KDIM + c0] = s0;
  psum [(size_t)b * KDIM + c1] = s1;
  psum2[(size_t)b * KDIM + c0] = q0;
  psum2[(size_t)b * KDIM + c1] = q1;
}

// ---------------- k2: reduce partials -> mean / invstd -----------------------
__global__ void k2_stats(const float* __restrict__ psum,
                         const float* __restrict__ psum2,
                         float* __restrict__ mean,
                         float* __restrict__ invstd) {
  const int k = blockIdx.x * blockDim.x + threadIdx.x;
  if (k >= KDIM) return;
  float s = 0.f, s2 = 0.f;
  for (int bb = 0; bb < NBLK; ++bb) {
    s  += psum [(size_t)bb * KDIM + k];
    s2 += psum2[(size_t)bb * KDIM + k];
  }
  const float m = s / (float)NROWS;
  const float v = s2 / (float)NROWS - m * m;
  mean[k]   = m;
  invstd[k] = rsqrtf(v + BN_EPS);
}

// ---------------- k3: in-place normalize (float4) ----------------------------
__global__ __launch_bounds__(256)
void k3_normalize(float* __restrict__ Y,
                  const float* __restrict__ mean,
                  const float* __restrict__ invstd) {
  const size_t e = ((size_t)blockIdx.x * blockDim.x + threadIdx.x) * 4;
  const int    k = (int)(e & (KDIM - 1));
  v4f y = *(v4f*)(Y + e);
  v4f m = *(const v4f*)(mean + k);
  v4f s = *(const v4f*)(invstd + k);
  v4f o = (y - m) * s;
  *(v4f*)(Y + e) = o;
}

// ---------------- host launcher ---------------------------------------------
extern "C" void kernel_launch(void* const* d_in, const int* in_sizes, int n_in,
                              void* d_out, int out_size, void* d_ws, size_t ws_size,
                              hipStream_t stream) {
  const float* X  = (const float*)d_in[0];
  const float* C1 = (const float*)d_in[1];
  const float* C2 = (const float*)d_in[2];
  const float* C3 = (const float*)d_in[3];
  float* Y = (float*)d_out;

  char* ws = (char*)d_ws;
  __bf16* C1T  = (__bf16*)ws;                                   // 512 KB
  float* psum  = (float*)(ws + (size_t)524288);                 // 2 MB
  float* psum2 = (float*)(ws + (size_t)524288 + 2097152);       // 2 MB
  float* mean  = (float*)(ws + (size_t)524288 + 2 * 2097152);   // 2 KB
  float* invst = mean + KDIM;                                   // 2 KB

  k0_convert_transpose<<<KDIM, 256, 0, stream>>>(C1, C1T);
  k1_quad_y_stats<<<NBLK, 256, 0, stream>>>(X, C1T, C2, C3, Y, psum, psum2);
  k2_stats<<<(KDIM + 255) / 256, 256, 0, stream>>>(psum, psum2, mean, invst);
  k3_normalize<<<(NROWS * KDIM) / (256 * 4), 256, 0, stream>>>(Y, mean, invst);
}
